// LCAOInteraction_12438225289735
// MI455X (gfx1250) — matrix operand order repaired and checked by reading, not compiled, not run on hardware
//
#include <hip/hip_runtime.h>
#include <hip/hip_bf16.h>
#include <math.h>

typedef __bf16 bf16_t;
typedef __attribute__((ext_vector_type(16))) __bf16 v16bf;
typedef __attribute__((ext_vector_type(8)))  float  v8f;

#define DEVINL __device__ __forceinline__

DEVINL float fast_sigmoid(float x) { return 1.0f / (1.0f + __expf(-x)); }
DEVINL float fast_silu(float x)    { return x / (1.0f + __expf(-x)); }

// 16-byte global -> LDS copy using the CDNA5 async-DMA path (ASYNCcnt).
DEVINL void copy16_g2l(const void* g, void* l) {
  unsigned lofs = (unsigned)(unsigned long long)(size_t)l;
  unsigned long long gaddr = (unsigned long long)(size_t)g;
  asm volatile("global_load_async_to_lds_b128 %0, %1, off"
               :: "v"(lofs), "v"(gaddr) : "memory");
}
DEVINL void g2l_wait() {
  asm volatile("s_wait_asynccnt 0x0" ::: "memory");
}

// ---- WMMA helpers (wave32, v_wmma_f32_16x16x32_bf16) --------------------

DEVINL v8f wmma_bf16(v16bf a, v16bf b, v8f c) {
  return __builtin_amdgcn_wmma_f32_16x16x32_bf16(
      false, a, false, b, (short)0, c, false, false);
}

// A fragment: 16x32 tile, LDS holds f32 row-major (stride elems); convert on read.
DEVINL v16bf load_frag_a_f32(const float* lds, int stride, int ks) {
  int lane = threadIdx.x & 31;
  int row  = lane & 15;
  int koff = (lane >> 4) ? 16 : 0;
  const float* p0 = lds + row * stride + ks + koff;
  v8f lo = *(const v8f*)(p0);
  v8f hi = *(const v8f*)(p0 + 8);
  v16bf r;
#pragma unroll
  for (int j = 0; j < 8; ++j) { r[j] = (bf16_t)lo[j]; r[8 + j] = (bf16_t)hi[j]; }
  return r;
}

// B fragment: weights stored TRANSPOSED bf16 in LDS: Wt[n*kdim + k] = W[k*N + n].
DEVINL v16bf load_frag_b(const bf16_t* Wt, int kdim, int col0, int ks) {
  int lane = threadIdx.x & 31;
  int col  = col0 + (lane & 15);
  int koff = (lane >> 4) ? 16 : 0;
  return *(const v16bf*)(Wt + col * kdim + ks + koff);
}

// Load K x Nn row-major f32 weights from global, store transposed bf16 in LDS.
DEVINL void load_weight_t(const float* __restrict__ W, bf16_t* Wt,
                          int K, int Nn, int tid, int nthr) {
  for (int i = tid; i < K * Nn; i += nthr) {
    int k = i / Nn, n = i - k * Nn;
    Wt[n * K + k] = (bf16_t)W[i];
  }
}

// ---- Kernel 1: h = x @ W_node + b ; x1 = h[:, :64], xk_sig = sigmoid(h[:,64:])
__global__ void __launch_bounds__(64)
k_node_mlp(const float* __restrict__ x, const float* __restrict__ Wn,
           const float* __restrict__ bn, float* __restrict__ x1,
           float* __restrict__ xk_sig, int N) {
  __shared__ __align__(32) bf16_t Wt[128 * 128];
  __shared__ __align__(32) float  Abuf[2][16 * 128];
  int tid = threadIdx.x, w = tid >> 5, lane = tid & 31;
  int r0 = blockIdx.x * 32 + w * 16;
  if (r0 < N) {
    const float* src = x + (size_t)r0 * 128;
    for (int i = lane; i < 512; i += 32) copy16_g2l(src + i * 4, Abuf[w] + i * 4);
  }
  load_weight_t(Wn, Wt, 128, 128, tid, 64);
  __syncthreads();
  if (r0 >= N) return;
  g2l_wait();
  int c = lane & 15, half = lane >> 4;
  v8f acc[8];
#pragma unroll
  for (int nt = 0; nt < 8; ++nt) {
    float bval = bn[nt * 16 + c];
#pragma unroll
    for (int r = 0; r < 8; ++r) acc[nt][r] = bval;
  }
#pragma unroll
  for (int ks = 0; ks < 128; ks += 32) {
    v16bf a = load_frag_a_f32(Abuf[w], 128, ks);
#pragma unroll
    for (int nt = 0; nt < 8; ++nt)
      acc[nt] = wmma_bf16(a, load_frag_b(Wt, 128, nt * 16, ks), acc[nt]);
  }
#pragma unroll
  for (int nt = 0; nt < 8; ++nt)
#pragma unroll
    for (int r = 0; r < 8; ++r) {
      size_t row = (size_t)(r0 + r + 8 * half);
      int col = nt * 16 + c;
      if (nt < 4) x1[row * 64 + col] = acc[nt][r];
      else        xk_sig[row * 64 + (col - 64)] = fast_sigmoid(acc[nt][r]);
    }
}

// ---- Kernel 2: cst2 = silu(silu(cst @ Wc1) @ Wc2), rows R = E*16, bf16 out
__global__ void __launch_bounds__(128)
k_cst_mlp(const float* __restrict__ cst, const float* __restrict__ Wc1,
          const float* __restrict__ Wc2, bf16_t* __restrict__ cst2,
          long long R) {
  __shared__ __align__(32) bf16_t Wt1[64 * 64];
  __shared__ __align__(32) bf16_t Wt2[64 * 64];
  __shared__ __align__(32) float  Abuf[4][16 * 64];
  int tid = threadIdx.x, w = tid >> 5, lane = tid & 31;
  long long r0 = (long long)blockIdx.x * 64 + w * 16;
  if (r0 < R) {
    const float* src = cst + r0 * 64;
    for (int i = lane; i < 256; i += 32) copy16_g2l(src + i * 4, Abuf[w] + i * 4);
  }
  load_weight_t(Wc1, Wt1, 64, 64, tid, 128);
  load_weight_t(Wc2, Wt2, 64, 64, tid, 128);
  __syncthreads();
  if (r0 >= R) return;
  g2l_wait();
  int c = lane & 15, half = lane >> 4;
  v8f t[4];
#pragma unroll
  for (int nt = 0; nt < 4; ++nt)
#pragma unroll
    for (int r = 0; r < 8; ++r) t[nt][r] = 0.0f;
#pragma unroll
  for (int ks = 0; ks < 64; ks += 32) {
    v16bf a = load_frag_a_f32(Abuf[w], 64, ks);
#pragma unroll
    for (int nt = 0; nt < 4; ++nt)
      t[nt] = wmma_bf16(a, load_frag_b(Wt1, 64, nt * 16, ks), t[nt]);
  }
#pragma unroll
  for (int nt = 0; nt < 4; ++nt)
#pragma unroll
    for (int r = 0; r < 8; ++r)
      Abuf[w][(r + 8 * half) * 64 + nt * 16 + c] = fast_silu(t[nt][r]);
#pragma unroll
  for (int nt = 0; nt < 4; ++nt)
#pragma unroll
    for (int r = 0; r < 8; ++r) t[nt][r] = 0.0f;
#pragma unroll
  for (int ks = 0; ks < 64; ks += 32) {
    v16bf a = load_frag_a_f32(Abuf[w], 64, ks);
#pragma unroll
    for (int nt = 0; nt < 4; ++nt)
      t[nt] = wmma_bf16(a, load_frag_b(Wt2, 64, nt * 16, ks), t[nt]);
  }
#pragma unroll
  for (int nt = 0; nt < 4; ++nt)
#pragma unroll
    for (int r = 0; r < 8; ++r)
      cst2[(r0 + r + 8 * half) * 64 + nt * 16 + c] = (bf16_t)fast_silu(t[nt][r]);
}

// ---- Kernel 3: zero fill
__global__ void k_zero(float* __restrict__ p, long long n) {
  long long i = (long long)blockIdx.x * blockDim.x + threadIdx.x;
  long long stride = (long long)gridDim.x * blockDim.x;
  for (; i < n; i += stride) p[i] = 0.0f;
}

// ---- Kernel 4: per-triplet three-body weight + normalize + gate + segsum
__global__ void __launch_bounds__(64)
k_triplet(const bf16_t* __restrict__ cst2, const float* __restrict__ rb,
          const float* __restrict__ shb, const int* __restrict__ tri_idx_k,
          const int* __restrict__ edge_idx_ks, const int* __restrict__ edge_idx_st,
          const float* __restrict__ xk_sig, float* __restrict__ tbw, int T) {
  int t = blockIdx.x;
  int h = threadIdx.x;  // 0..63
  __shared__ __align__(16) bf16_t cbuf[16 * 64];
  __shared__ float p[16];
  __shared__ float red[64];
  int eks = edge_idx_ks[t];
  const bf16_t* cb = cst2 + (size_t)eks * 16 * 64;
  // stage 2KB tile via async DMA (2 chunks/thread)
  for (int i = h; i < 128; i += 64)
    copy16_g2l((const char*)cb + i * 16, (char*)cbuf + i * 16);
  if (h < 16) p[h] = rb[(size_t)eks * 16 + h] * shb[(size_t)t * 16 + h];
  g2l_wait();
  __syncthreads();
  float s = 0.0f;
#pragma unroll
  for (int d = 0; d < 16; ++d) s += p[d] * (float)cbuf[d * 64 + h];
  red[h] = s * s;
  __syncthreads();
  for (int off = 32; off > 0; off >>= 1) {
    if (h < off) red[h] += red[h + off];
    __syncthreads();
  }
  float w = s / fmaxf(sqrtf(red[0]), 1e-12f);
  w *= xk_sig[(size_t)tri_idx_k[t] * 64 + h];
  atomicAdd(&tbw[(size_t)edge_idx_st[t] * 64 + h], w);
}

// ---- Kernel 5: Out = Ain @ W  (rows x 64 @ 64 x 64), plain WMMA GEMM
__global__ void __launch_bounds__(128)
k_gemm64(const float* __restrict__ Ain, const float* __restrict__ W,
         float* __restrict__ Out, long long rows) {
  __shared__ __align__(32) bf16_t Wt[64 * 64];
  __shared__ __align__(32) float  Abuf[4][16 * 64];
  int tid = threadIdx.x, w = tid >> 5, lane = tid & 31;
  long long r0 = (long long)blockIdx.x * 64 + w * 16;
  if (r0 < rows) {
    const float* src = Ain + r0 * 64;
    for (int i = lane; i < 256; i += 32) copy16_g2l(src + i * 4, Abuf[w] + i * 4);
  }
  load_weight_t(W, Wt, 64, 64, tid, 128);
  __syncthreads();
  if (r0 >= rows) return;
  g2l_wait();
  int c = lane & 15, half = lane >> 4;
  v8f t[4];
#pragma unroll
  for (int nt = 0; nt < 4; ++nt)
#pragma unroll
    for (int r = 0; r < 8; ++r) t[nt][r] = 0.0f;
#pragma unroll
  for (int ks = 0; ks < 64; ks += 32) {
    v16bf a = load_frag_a_f32(Abuf[w], 64, ks);
#pragma unroll
    for (int nt = 0; nt < 4; ++nt)
      t[nt] = wmma_bf16(a, load_frag_b(Wt, 64, nt * 16, ks), t[nt]);
  }
#pragma unroll
  for (int nt = 0; nt < 4; ++nt)
#pragma unroll
    for (int r = 0; r < 8; ++r)
      Out[(r0 + r + 8 * half) * 64 + nt * 16 + c] = t[nt][r];
}

// ---- Kernel 6: lcao = normalize( (sum_d rb*cst2) * (1 + tbw3) )
__global__ void __launch_bounds__(64)
k_lcao(const bf16_t* __restrict__ cst2, const float* __restrict__ rb,
       const float* __restrict__ tbw3, float* __restrict__ lcao, int E) {
  int e = blockIdx.x;
  int h = threadIdx.x;
  __shared__ __align__(16) bf16_t cbuf[16 * 64];
  __shared__ float p[16];
  __shared__ float red[64];
  const bf16_t* cb = cst2 + (size_t)e * 16 * 64;
  for (int i = h; i < 128; i += 64)
    copy16_g2l((const char*)cb + i * 16, (char*)cbuf + i * 16);
  if (h < 16) p[h] = rb[(size_t)e * 16 + h];
  g2l_wait();
  __syncthreads();
  float s = 0.0f;
#pragma unroll
  for (int d = 0; d < 16; ++d) s += p[d] * (float)cbuf[d * 64 + h];
  float v = s * (1.0f + tbw3[(size_t)e * 64 + h]);
  red[h] = v * v;
  __syncthreads();
  for (int off = 32; off > 0; off >>= 1) {
    if (h < off) red[h] += red[h + off];
    __syncthreads();
  }
  lcao[(size_t)e * 64 + h] = v / fmaxf(sqrtf(red[0]), 1e-12f);
}

// ---- Kernel 7: node-feat MLP (gathered) * (lcao @ W_basis) -> atomic agg
__global__ void __launch_bounds__(64)
k_edge_msg(const float* __restrict__ x1, const float* __restrict__ lcao,
           const int* __restrict__ idx_s, const int* __restrict__ idx_t,
           const float* __restrict__ Wn1, const float* __restrict__ bn1,
           const float* __restrict__ Wn2, const float* __restrict__ bn2,
           const float* __restrict__ Wb, float* __restrict__ agg, int E) {
  __shared__ __align__(32) bf16_t Wt1[64 * 128];  // (n=64) x (k=128)
  __shared__ __align__(32) bf16_t Wt2[64 * 64];
  __shared__ __align__(32) bf16_t Wtb[64 * 64];
  __shared__ __align__(32) float  Abuf[2][16 * 128];
  __shared__ float nf[2][16 * 64];
  int tid = threadIdx.x, w = tid >> 5, lane = tid & 31;
  long long e0 = (long long)blockIdx.x * 32 + w * 16;
  if (e0 < E) {
    for (int i = lane; i < 512; i += 32) {
      int r = i >> 5, ch = i & 31;
      int half2 = ch >> 4;
      int cc = (ch & 15) * 4;
      long long e = e0 + r;
      int node = half2 ? idx_t[e] : idx_s[e];
      copy16_g2l(x1 + (size_t)node * 64 + cc, Abuf[w] + r * 128 + half2 * 64 + cc);
    }
  }
  load_weight_t(Wn1, Wt1, 128, 64, tid, 64);
  load_weight_t(Wn2, Wt2, 64, 64, tid, 64);
  load_weight_t(Wb,  Wtb, 64, 64, tid, 64);
  __syncthreads();
  if (e0 >= E) return;
  g2l_wait();
  int c = lane & 15, half = lane >> 4;
  v8f t[4];
  // MLP layer 1: (16x128) @ (128x64) + b1, silu
#pragma unroll
  for (int nt = 0; nt < 4; ++nt) {
    float bval = bn1[nt * 16 + c];
#pragma unroll
    for (int r = 0; r < 8; ++r) t[nt][r] = bval;
  }
#pragma unroll
  for (int ks = 0; ks < 128; ks += 32) {
    v16bf a = load_frag_a_f32(Abuf[w], 128, ks);
#pragma unroll
    for (int nt = 0; nt < 4; ++nt)
      t[nt] = wmma_bf16(a, load_frag_b(Wt1, 128, nt * 16, ks), t[nt]);
  }
#pragma unroll
  for (int nt = 0; nt < 4; ++nt)
#pragma unroll
    for (int r = 0; r < 8; ++r)
      Abuf[w][(r + 8 * half) * 64 + nt * 16 + c] = fast_silu(t[nt][r]);
  // MLP layer 2: (16x64) @ (64x64) + b2, silu -> nf
#pragma unroll
  for (int nt = 0; nt < 4; ++nt) {
    float bval = bn2[nt * 16 + c];
#pragma unroll
    for (int r = 0; r < 8; ++r) t[nt][r] = bval;
  }
#pragma unroll
  for (int ks = 0; ks < 64; ks += 32) {
    v16bf a = load_frag_a_f32(Abuf[w], 64, ks);
#pragma unroll
    for (int nt = 0; nt < 4; ++nt)
      t[nt] = wmma_bf16(a, load_frag_b(Wt2, 64, nt * 16, ks), t[nt]);
  }
#pragma unroll
  for (int nt = 0; nt < 4; ++nt)
#pragma unroll
    for (int r = 0; r < 8; ++r)
      nf[w][(r + 8 * half) * 64 + nt * 16 + c] = fast_silu(t[nt][r]);
  // stage lcao rows as A (per-wave region; contiguous 16x64 f32)
  {
    const float* src = lcao + e0 * 64;
    for (int i = lane; i < 256; i += 32) copy16_g2l(src + i * 4, Abuf[w] + i * 4);
    g2l_wait();
  }
  // lb = lcao @ W_basis ; msg = lb * nf ; scatter-add to agg[idx_s]
#pragma unroll
  for (int nt = 0; nt < 4; ++nt)
#pragma unroll
    for (int r = 0; r < 8; ++r) t[nt][r] = 0.0f;
#pragma unroll
  for (int ks = 0; ks < 64; ks += 32) {
    v16bf a = load_frag_a_f32(Abuf[w], 64, ks);
#pragma unroll
    for (int nt = 0; nt < 4; ++nt)
      t[nt] = wmma_bf16(a, load_frag_b(Wtb, 64, nt * 16, ks), t[nt]);
  }
#pragma unroll
  for (int nt = 0; nt < 4; ++nt)
#pragma unroll
    for (int r = 0; r < 8; ++r) {
      long long e = e0 + r + 8 * half;
      int col = nt * 16 + c;
      float m = t[nt][r] * nf[w][(r + 8 * half) * 64 + col];
      atomicAdd(&agg[(size_t)idx_s[e] * 64 + col], m);
    }
}

// ---- Kernel 8: out = x + agg @ W_out  (N x 64 @ 64 x 128)
__global__ void __launch_bounds__(128)
k_final(const float* __restrict__ x, const float* __restrict__ agg,
        const float* __restrict__ Wout, float* __restrict__ out, int N) {
  __shared__ __align__(32) bf16_t Wt[128 * 64];  // (n=128) x (k=64)
  __shared__ __align__(32) float  Abuf[4][16 * 64];
  int tid = threadIdx.x, w = tid >> 5, lane = tid & 31;
  int r0 = blockIdx.x * 64 + w * 16;
  if (r0 < N) {
    const float* src = agg + (size_t)r0 * 64;
    for (int i = lane; i < 256; i += 32) copy16_g2l(src + i * 4, Abuf[w] + i * 4);
  }
  load_weight_t(Wout, Wt, 64, 128, tid, 128);
  __syncthreads();
  if (r0 >= N) return;
  g2l_wait();
  int c = lane & 15, half = lane >> 4;
  v8f acc[8];
#pragma unroll
  for (int nt = 0; nt < 8; ++nt)
#pragma unroll
    for (int r = 0; r < 8; ++r) acc[nt][r] = 0.0f;
#pragma unroll
  for (int ks = 0; ks < 64; ks += 32) {
    v16bf a = load_frag_a_f32(Abuf[w], 64, ks);
#pragma unroll
    for (int nt = 0; nt < 8; ++nt)
      acc[nt] = wmma_bf16(a, load_frag_b(Wt, 64, nt * 16, ks), acc[nt]);
  }
#pragma unroll
  for (int nt = 0; nt < 8; ++nt)
#pragma unroll
    for (int r = 0; r < 8; ++r) {
      size_t row = (size_t)(r0 + r + 8 * half);
      int col = nt * 16 + c;
      out[row * 128 + col] = x[row * 128 + col] + acc[nt][r];
    }
}

// ---- Host launcher -------------------------------------------------------

extern "C" void kernel_launch(void* const* d_in, const int* in_sizes, int n_in,
                              void* d_out, int out_size, void* d_ws, size_t ws_size,
                              hipStream_t stream) {
  const float* x       = (const float*)d_in[0];
  const float* cst     = (const float*)d_in[1];
  const float* rb      = (const float*)d_in[2];
  const float* shb     = (const float*)d_in[3];
  const int*   idx_s   = (const int*)d_in[4];
  const int*   idx_t   = (const int*)d_in[5];
  const int*   tri_k   = (const int*)d_in[6];
  const int*   e_ks    = (const int*)d_in[7];
  const int*   e_st    = (const int*)d_in[8];
  const float* W_node  = (const float*)d_in[9];
  const float* b_node  = (const float*)d_in[10];
  const float* W_c1    = (const float*)d_in[11];
  const float* W_c2    = (const float*)d_in[12];
  const float* W_three = (const float*)d_in[13];
  const float* W_basis = (const float*)d_in[14];
  const float* W_n1    = (const float*)d_in[15];
  const float* b_n1    = (const float*)d_in[16];
  const float* W_n2    = (const float*)d_in[17];
  const float* b_n2    = (const float*)d_in[18];
  const float* W_out   = (const float*)d_in[19];
  float* out = (float*)d_out;

  const int N = in_sizes[0] / 128;
  const int E = in_sizes[2] / 16;
  const int T = in_sizes[3] / 16;
  const long long R = (long long)E * 16;

  char* wsb = (char*)d_ws;
  size_t off = 0;
  auto take = [&](size_t bytes) -> void* {
    void* p = wsb + off;
    off = (off + bytes + 255) & ~(size_t)255;
    return p;
  };
  float*  x1     = (float*)take((size_t)N * 64 * 4);
  float*  xk_sig = (float*)take((size_t)N * 64 * 4);
  bf16_t* cst2   = (bf16_t*)take((size_t)R * 64 * 2);
  float*  tbw    = (float*)take((size_t)E * 64 * 4);
  float*  tbw3   = (float*)take((size_t)E * 64 * 4);
  float*  lcao   = (float*)take((size_t)E * 64 * 4);
  float*  agg    = (float*)take((size_t)N * 64 * 4);
  (void)ws_size; (void)n_in; (void)out_size;

  k_node_mlp<<<(N + 31) / 32, 64, 0, stream>>>(x, W_node, b_node, x1, xk_sig, N);
  k_cst_mlp<<<(int)((R + 63) / 64), 128, 0, stream>>>(cst, W_c1, W_c2, cst2, R);
  k_zero<<<2048, 256, 0, stream>>>(tbw, (long long)E * 64);
  k_zero<<<512, 256, 0, stream>>>(agg, (long long)N * 64);
  k_triplet<<<T, 64, 0, stream>>>(cst2, rb, shb, tri_k, e_ks, e_st, xk_sig, tbw, T);
  k_gemm64<<<(E + 63) / 64, 128, 0, stream>>>(tbw, W_three, tbw3, (long long)E);
  k_lcao<<<E, 64, 0, stream>>>(cst2, rb, tbw3, lcao, E);
  k_edge_msg<<<(E + 31) / 32, 64, 0, stream>>>(x1, lcao, idx_s, idx_t, W_n1, b_n1,
                                               W_n2, b_n2, W_basis, agg, E);
  k_final<<<(N + 63) / 64, 128, 0, stream>>>(x, agg, W_out, out, N);
}
